// PMNQP_43344809951259
// MI455X (gfx1250) — matrix operand order repaired and checked by reading, not compiled, not run on hardware
//
#include <hip/hip_runtime.h>
#include <stdint.h>
#include <stddef.h>

// ---------------------------------------------------------------------------
// Problem constants (from reference)
// ---------------------------------------------------------------------------
#define DIMD      768
#define MEMLEN    64
#define NATTRS    400
#define NOBJS     500
#define NBATCH    2048

typedef __attribute__((ext_vector_type(16))) __bf16 bf16x16;
typedef __attribute__((ext_vector_type(8)))  float  f32x8;

__device__ __forceinline__ unsigned short f2bf(float f) {
  unsigned int u = __float_as_uint(f);
  u += 0x7FFFu + ((u >> 16) & 1u);          // round to nearest even
  return (unsigned short)(u >> 16);
}
__device__ __forceinline__ float bf2f(unsigned short h) {
  return __uint_as_float(((unsigned int)h) << 16);
}

// ---------------------------------------------------------------------------
// f32 -> bf16 conversion
// ---------------------------------------------------------------------------
__global__ void k_cvt_bf16(const float* __restrict__ in,
                           unsigned short* __restrict__ out, int n) {
  int i = blockIdx.x * blockDim.x + threadIdx.x;
  if (i < n) out[i] = f2bf(in[i]);
}

// ---------------------------------------------------------------------------
// Generic bf16 WMMA GEMM:  C[m,n] = sum_k A[m,k] * W[n,k] + bias[n] (+ R[m,n])
//   A : [M,K] bf16 row-major, W : [N,K] bf16 row-major (i.e. computes A @ W^T)
//   Output either fp32 (Cf) or bf16 (Cb).
//   One wave computes a 32(M) x 64(N) tile: 2 A fragments reuse 4 B fragments
//   per K-step -> 8 v_wmma per 32-wide K-step, 2x the FLOP/byte of a 16x64
//   tile. blockDim = 256 (8 waves).
//   M must be a multiple of 32, N a multiple of 64, K a multiple of 32.
// ---------------------------------------------------------------------------
union AFragU { bf16x16 v; uint4 q[2]; };

__global__ void __launch_bounds__(256)
k_gemm_bf16(const unsigned short* __restrict__ A,
            const unsigned short* __restrict__ W,
            const float* __restrict__ bias,
            const float* __restrict__ residual,   // may be null
            float* __restrict__ Cf,               // may be null
            unsigned short* __restrict__ Cb,      // may be null
            int M, int N, int K) {
  const int wave = threadIdx.x >> 5;
  const int lane = threadIdx.x & 31;
  const int l    = lane & 15;
  const int half = lane >> 4;
  const int nGroups = N >> 6;
  const int totalTiles = (M >> 5) * nGroups;
  const int tile = blockIdx.x * 8 + wave;
  if (tile >= totalTiles) return;
  const int mBase = (tile / nGroups) << 5;
  const int nBase = (tile % nGroups) << 6;

  f32x8 acc[2][4] = {};

  // A fragments: rows mBase+l and mBase+16+l; lane-half 0 takes K {0..7,16..23},
  // half 1 takes K {8..15,24..31} (ISA 16-bit A layout).
  const unsigned short* a0 = A + (size_t)(mBase + l) * K + (half ? 8 : 0);
  const unsigned short* a1 = a0 + (size_t)16 * K;
  // B fragments: column = nBase + nt*16 + l; same K split as A (half 0: {0..7,16..23}, half 1: {8..15,24..31}).
  const unsigned short* b0 = W + (size_t)(nBase + l) * K + (half ? 8 : 0);
  const unsigned short* b1 = b0 + (size_t)16 * K;
  const unsigned short* b2 = b0 + (size_t)32 * K;
  const unsigned short* b3 = b0 + (size_t)48 * K;

#pragma unroll 2
  for (int kb = 0; kb < K; kb += 32) {
    AFragU fa0, fa1, w0, w1, w2, w3;
    fa0.q[0] = *(const uint4*)(a0 + kb);
    fa0.q[1] = *(const uint4*)(a0 + kb + 16);
    fa1.q[0] = *(const uint4*)(a1 + kb);
    fa1.q[1] = *(const uint4*)(a1 + kb + 16);
    w0.q[0]  = *(const uint4*)(b0 + kb);
    w0.q[1]  = *(const uint4*)(b0 + kb + 16);
    w1.q[0]  = *(const uint4*)(b1 + kb);
    w1.q[1]  = *(const uint4*)(b1 + kb + 16);
    w2.q[0]  = *(const uint4*)(b2 + kb);
    w2.q[1]  = *(const uint4*)(b2 + kb + 16);
    w3.q[0]  = *(const uint4*)(b3 + kb);
    w3.q[1]  = *(const uint4*)(b3 + kb + 16);
    acc[0][0] = __builtin_amdgcn_wmma_f32_16x16x32_bf16(false, fa0.v, false, w0.v, (short)0, acc[0][0], false, false);
    acc[0][1] = __builtin_amdgcn_wmma_f32_16x16x32_bf16(false, fa0.v, false, w1.v, (short)0, acc[0][1], false, false);
    acc[0][2] = __builtin_amdgcn_wmma_f32_16x16x32_bf16(false, fa0.v, false, w2.v, (short)0, acc[0][2], false, false);
    acc[0][3] = __builtin_amdgcn_wmma_f32_16x16x32_bf16(false, fa0.v, false, w3.v, (short)0, acc[0][3], false, false);
    acc[1][0] = __builtin_amdgcn_wmma_f32_16x16x32_bf16(false, fa1.v, false, w0.v, (short)0, acc[1][0], false, false);
    acc[1][1] = __builtin_amdgcn_wmma_f32_16x16x32_bf16(false, fa1.v, false, w1.v, (short)0, acc[1][1], false, false);
    acc[1][2] = __builtin_amdgcn_wmma_f32_16x16x32_bf16(false, fa1.v, false, w2.v, (short)0, acc[1][2], false, false);
    acc[1][3] = __builtin_amdgcn_wmma_f32_16x16x32_bf16(false, fa1.v, false, w3.v, (short)0, acc[1][3], false, false);
  }

#pragma unroll
  for (int mt = 0; mt < 2; ++mt) {
#pragma unroll
    for (int nt = 0; nt < 4; ++nt) {
      const int col = nBase + nt * 16 + l;
      const float bv = bias[col];
#pragma unroll
      for (int g = 0; g < 8; ++g) {
        const int row = mBase + mt * 16 + g + (half << 3);
        const size_t idx = (size_t)row * N + col;
        float v = acc[mt][nt][g] + bv;
        if (residual) v += residual[idx];
        if (Cf) Cf[idx] = v;
        if (Cb) Cb[idx] = f2bf(v);
      }
    }
  }
}

// ---------------------------------------------------------------------------
// Attention core: per-sample scores -> softmax -> context.
//   q     : [B,768] fp32 (already projected + bias)
//   Kbank : [C,64,768] bf16, Vbank : [C,64,768] bf16
//   ctx   : [B,768] bf16 (input to output projection)
// One block (256 threads) per sample.
// ---------------------------------------------------------------------------
__global__ void __launch_bounds__(256)
k_attn_core(const float* __restrict__ q, const int* __restrict__ labels,
            const unsigned short* __restrict__ Kbank,
            const unsigned short* __restrict__ Vbank,
            unsigned short* __restrict__ ctx, int B) {
  __shared__ float sq[DIMD];
  __shared__ float spart[MEMLEN][4];
  __shared__ float sscore[MEMLEN];
  __shared__ float sinv;

  const int b = blockIdx.x;
  const int t = threadIdx.x;
  if (b >= B) return;

  const float* qb = q + (size_t)b * DIMD;
  for (int i = t; i < DIMD; i += 256) sq[i] = qb[i];
  __syncthreads();

  const int c = labels[b];
  const unsigned short* Kc = Kbank + (size_t)c * MEMLEN * DIMD;
  const unsigned short* Vc = Vbank + (size_t)c * MEMLEN * DIMD;

  // scores: 4 threads per slot, 192 elems each (vectorized 4 bf16 per step)
  {
    const int m = t >> 2, p = t & 3;
    const unsigned short* kr = Kc + (size_t)m * DIMD + p * 192;
    const float* qs = sq + p * 192;
    float s = 0.f;
    for (int i = 0; i < 192; i += 4) {
      uint2 kk = *(const uint2*)(kr + i);
      s += bf2f((unsigned short)(kk.x & 0xFFFFu)) * qs[i + 0];
      s += bf2f((unsigned short)(kk.x >> 16))     * qs[i + 1];
      s += bf2f((unsigned short)(kk.y & 0xFFFFu)) * qs[i + 2];
      s += bf2f((unsigned short)(kk.y >> 16))     * qs[i + 3];
    }
    spart[m][p] = s;
  }
  __syncthreads();
  if (t < MEMLEN) {
    const float scale = rsqrtf((float)DIMD);
    sscore[t] = (spart[t][0] + spart[t][1] + spart[t][2] + spart[t][3]) * scale;
  }
  __syncthreads();
  if (t == 0) {
    float mx = sscore[0];
    for (int m = 1; m < MEMLEN; ++m) mx = fmaxf(mx, sscore[m]);
    float sum = 0.f;
    for (int m = 0; m < MEMLEN; ++m) { float e = __expf(sscore[m] - mx); sscore[m] = e; sum += e; }
    sinv = 1.0f / sum;
  }
  __syncthreads();

  const float inv = sinv;
  for (int d = t; d < DIMD; d += 256) {
    float a = 0.f;
    for (int m = 0; m < MEMLEN; ++m) a += sscore[m] * bf2f(Vc[(size_t)m * DIMD + d]);
    ctx[(size_t)b * DIMD + d] = f2bf(a * inv);
  }
}

// ---------------------------------------------------------------------------
// Memory update: last batch index with a given label wins (including no-op
// writes), worst slot & threshold taken from the ORIGINAL entropy bank.
// ---------------------------------------------------------------------------
__global__ void k_winner_init(int* __restrict__ w, int C) {
  int i = blockIdx.x * blockDim.x + threadIdx.x;
  if (i < C) w[i] = -1;
}
__global__ void k_winner_scan(const int* __restrict__ labels, int* __restrict__ w, int B) {
  int i = blockIdx.x * blockDim.x + threadIdx.x;
  if (i < B) atomicMax(&w[labels[i]], i);
}
__global__ void __launch_bounds__(256)
k_apply_update(const float* __restrict__ x, const float* __restrict__ ent,
               const float* __restrict__ bank, const int* __restrict__ winner,
               float* __restrict__ new_mem, float* __restrict__ new_bank, int C) {
  __shared__ int s_do, s_w, s_worst;
  __shared__ float s_val;
  const int c = blockIdx.x;
  if (c >= C) return;
  if (threadIdx.x == 0) {
    s_do = 0;
    const int w = winner[c];
    if (w >= 0) {
      const float* br = bank + (size_t)c * MEMLEN;
      int worst = 0; float mx = br[0];
      for (int m = 1; m < MEMLEN; ++m)
        if (br[m] > mx) { mx = br[m]; worst = m; }   // first-max tie-break (argmax)
      if (ent[w] < mx) { s_do = 1; s_w = w; s_worst = worst; s_val = ent[w]; }
    }
  }
  __syncthreads();
  if (s_do) {
    float* dst = new_mem + ((size_t)c * MEMLEN + s_worst) * DIMD;
    const float* src = x + (size_t)s_w * DIMD;
    for (int d = threadIdx.x; d < DIMD; d += 256) dst[d] = src[d];
    if (threadIdx.x == 0) new_bank[(size_t)c * MEMLEN + s_worst] = s_val;
  }
}

// ---------------------------------------------------------------------------
// Host launcher
// ---------------------------------------------------------------------------
static inline size_t alignUp256(size_t x) { return (x + 255) & ~(size_t)255; }

extern "C" void kernel_launch(void* const* d_in, const int* in_sizes, int n_in,
                              void* d_out, int out_size, void* d_ws, size_t ws_size,
                              hipStream_t stream) {
  (void)in_sizes; (void)n_in; (void)out_size; (void)ws_size;

  const float* x_attr  = (const float*)d_in[0];
  const float* x_obj   = (const float*)d_in[1];
  const float* ent_a   = (const float*)d_in[2];
  const float* ent_o   = (const float*)d_in[3];
  const int*   lab_a   = (const int*)d_in[4];
  const int*   lab_o   = (const int*)d_in[5];
  const float* mem_a   = (const float*)d_in[6];
  const float* mem_o   = (const float*)d_in[7];
  const float* bank_a  = (const float*)d_in[8];
  const float* bank_o  = (const float*)d_in[9];
  const float* Wq_a = (const float*)d_in[10]; const float* bq_a = (const float*)d_in[11];
  const float* Wk_a = (const float*)d_in[12]; const float* bk_a = (const float*)d_in[13];
  const float* Wv_a = (const float*)d_in[14]; const float* bv_a = (const float*)d_in[15];
  const float* Wo_a = (const float*)d_in[16]; const float* bo_a = (const float*)d_in[17];
  const float* Wq_o = (const float*)d_in[18]; const float* bq_o = (const float*)d_in[19];
  const float* Wk_o = (const float*)d_in[20]; const float* bk_o = (const float*)d_in[21];
  const float* Wv_o = (const float*)d_in[22]; const float* bv_o = (const float*)d_in[23];
  const float* Wo_o = (const float*)d_in[24]; const float* bo_o = (const float*)d_in[25];

  // d_out layout (return order): out_a, out_o, new_am, new_om, new_ab, new_ob
  float* out = (float*)d_out;
  float* out_a  = out;
  float* out_o  = out_a + (size_t)NBATCH * DIMD;
  float* new_am = out_o + (size_t)NBATCH * DIMD;
  float* new_om = new_am + (size_t)NATTRS * MEMLEN * DIMD;
  float* new_ab = new_om + (size_t)NOBJS * MEMLEN * DIMD;
  float* new_ob = new_ab + (size_t)NATTRS * MEMLEN;

  // workspace carve
  char* p = (char*)d_ws;
  auto alloc = [&](size_t bytes) -> void* { void* r = (void*)p; p += alignUp256(bytes); return r; };
  const size_t nMemA = (size_t)NATTRS * MEMLEN * DIMD;   // 19,660,800
  const size_t nMemO = (size_t)NOBJS  * MEMLEN * DIMD;   // 24,576,000
  const size_t nX    = (size_t)NBATCH * DIMD;            // 1,572,864
  const size_t nW    = (size_t)DIMD * DIMD;              // 589,824

  unsigned short* memA_bf = (unsigned short*)alloc(nMemA * 2);
  unsigned short* memO_bf = (unsigned short*)alloc(nMemO * 2);
  unsigned short* xA_bf   = (unsigned short*)alloc(nX * 2);
  unsigned short* xO_bf   = (unsigned short*)alloc(nX * 2);
  unsigned short* Wbf[8];
  for (int i = 0; i < 8; ++i) Wbf[i] = (unsigned short*)alloc(nW * 2);
  unsigned short* Ka_bank = (unsigned short*)alloc(nMemA * 2);
  unsigned short* Va_bank = (unsigned short*)alloc(nMemA * 2);
  unsigned short* Ko_bank = (unsigned short*)alloc(nMemO * 2);
  unsigned short* Vo_bank = (unsigned short*)alloc(nMemO * 2);
  float* qA = (float*)alloc(nX * 4);
  float* qO = (float*)alloc(nX * 4);
  unsigned short* ctxA_bf = (unsigned short*)alloc(nX * 2);
  unsigned short* ctxO_bf = (unsigned short*)alloc(nX * 2);
  int* winA = (int*)alloc(NATTRS * 4);
  int* winO = (int*)alloc(NOBJS * 4);

  // ---- conversions to bf16 ------------------------------------------------
  auto cvt = [&](const float* src, unsigned short* dst, size_t n) {
    k_cvt_bf16<<<(unsigned)((n + 255) / 256), 256, 0, stream>>>(src, dst, (int)n);
  };
  cvt(mem_a, memA_bf, nMemA);
  cvt(mem_o, memO_bf, nMemO);
  cvt(x_attr, xA_bf, nX);
  cvt(x_obj,  xO_bf, nX);
  const float* Wsrc[8] = {Wq_a, Wk_a, Wv_a, Wo_a, Wq_o, Wk_o, Wv_o, Wo_o};
  for (int i = 0; i < 8; ++i) cvt(Wsrc[i], Wbf[i], nW);

  // ---- memory update path (independent of attention) ----------------------
  hipMemcpyAsync(new_am, mem_a,  nMemA * 4, hipMemcpyDeviceToDevice, stream);
  hipMemcpyAsync(new_om, mem_o,  nMemO * 4, hipMemcpyDeviceToDevice, stream);
  hipMemcpyAsync(new_ab, bank_a, (size_t)NATTRS * MEMLEN * 4, hipMemcpyDeviceToDevice, stream);
  hipMemcpyAsync(new_ob, bank_o, (size_t)NOBJS  * MEMLEN * 4, hipMemcpyDeviceToDevice, stream);
  k_winner_init<<<(NATTRS + 255) / 256, 256, 0, stream>>>(winA, NATTRS);
  k_winner_init<<<(NOBJS  + 255) / 256, 256, 0, stream>>>(winO, NOBJS);
  k_winner_scan<<<(NBATCH + 255) / 256, 256, 0, stream>>>(lab_a, winA, NBATCH);
  k_winner_scan<<<(NBATCH + 255) / 256, 256, 0, stream>>>(lab_o, winO, NBATCH);
  k_apply_update<<<NATTRS, 256, 0, stream>>>(x_attr, ent_a, bank_a, winA, new_am, new_ab, NATTRS);
  k_apply_update<<<NOBJS,  256, 0, stream>>>(x_obj,  ent_o, bank_o, winO, new_om, new_ob, NOBJS);

  // ---- GEMMs (32x64 per-wave tiles) ---------------------------------------
  auto gemm = [&](const unsigned short* A, const unsigned short* W, const float* bias,
                  const float* residual, float* Cf, unsigned short* Cb, int M) {
    const int tiles = (M / 32) * (DIMD / 64);
    k_gemm_bf16<<<(tiles + 7) / 8, 256, 0, stream>>>(A, W, bias, residual, Cf, Cb, M, DIMD, DIMD);
  };
  // q projections (fp32 out)
  gemm(xA_bf, Wbf[0], bq_a, nullptr, qA, nullptr, NBATCH);
  gemm(xO_bf, Wbf[4], bq_o, nullptr, qO, nullptr, NBATCH);
  // K / V class-bank projections (bf16 out) — whole banks, not gathered rows
  gemm(memA_bf, Wbf[1], bk_a, nullptr, nullptr, Ka_bank, NATTRS * MEMLEN);
  gemm(memA_bf, Wbf[2], bv_a, nullptr, nullptr, Va_bank, NATTRS * MEMLEN);
  gemm(memO_bf, Wbf[5], bk_o, nullptr, nullptr, Ko_bank, NOBJS * MEMLEN);
  gemm(memO_bf, Wbf[6], bv_o, nullptr, nullptr, Vo_bank, NOBJS * MEMLEN);

  // ---- attention core -----------------------------------------------------
  k_attn_core<<<NBATCH, 256, 0, stream>>>(qA, lab_a, Ka_bank, Va_bank, ctxA_bf, NBATCH);
  k_attn_core<<<NBATCH, 256, 0, stream>>>(qO, lab_o, Ko_bank, Vo_bank, ctxO_bf, NBATCH);

  // ---- output projection + residual add into d_out ------------------------
  gemm(ctxA_bf, Wbf[3], bo_a, x_attr, out_a, nullptr, NBATCH);
  gemm(ctxO_bf, Wbf[7], bo_o, x_obj,  out_o, nullptr, NBATCH);
}
